// MoEFFN_52810917871829
// MI455X (gfx1250) — matrix operand order repaired and checked by reading, not compiled, not run on hardware
//
#include <hip/hip_runtime.h>
#include <hip/hip_bf16.h>
#include <math.h>

// Problem constants (match reference)
#define D_EMBD 1024
#define FF_DIM 4096
#define N_EXP  8
#define T_TOK  2048

// GEMM tiling
#define BM 128
#define BN 128
#define BK 32
#define BKP 40   // padded k-stride (bf16 elems) for transposed B tile: 80B rows, 16B aligned

typedef __attribute__((ext_vector_type(16))) __bf16 v16bf;
typedef __attribute__((ext_vector_type(8)))  __bf16 v8bf;
typedef __attribute__((ext_vector_type(8)))  float  v8f;
typedef __attribute__((ext_vector_type(4)))  int    v4i;

union AB16 { v16bf v; v8bf h[2]; };

// Async global->LDS (CDNA5): use builtin if this toolchain declares it.
#if defined(__has_builtin)
# if __has_builtin(__builtin_amdgcn_global_load_async_to_lds_b128)
#  define MOE_ASYNC_LDS 1
# endif
#endif
#ifndef MOE_ASYNC_LDS
# define MOE_ASYNC_LDS 0
#endif

#if MOE_ASYNC_LDS
// Builtin expects: (int4 __device__* src, int4 __shared__* dst, int off, int cpol)
typedef __attribute__((address_space(1))) v4i as1_v4i;
typedef __attribute__((address_space(3))) v4i as3_v4i;
#endif

__device__ __forceinline__ void moe_wait_async() {
#if MOE_ASYNC_LDS
# if defined(__has_builtin) && __has_builtin(__builtin_amdgcn_s_wait_asynccnt)
    __builtin_amdgcn_s_wait_asynccnt(0);
# else
    asm volatile("s_wait_asynccnt 0x0" ::: "memory");
# endif
#endif
}

// ---------------------------------------------------------------------------
// Zero output + per-expert counters
// ---------------------------------------------------------------------------
__global__ void moe_zero_kernel(float* __restrict__ out, int* __restrict__ count) {
    int i = blockIdx.x * blockDim.x + threadIdx.x;
    if (i < T_TOK * D_EMBD) out[i] = 0.0f;
    if (i < N_EXP) count[i] = 0;
}

// ---------------------------------------------------------------------------
// Router: logits = x @ Wr + br ; softmax ; top-2 ; renormalize
// ---------------------------------------------------------------------------
__global__ __launch_bounds__(256)
void moe_router_kernel(const float* __restrict__ x, const float* __restrict__ Wr,
                       const float* __restrict__ br,
                       int* __restrict__ tok_e, float* __restrict__ tok_w,
                       int* __restrict__ count) {
    __shared__ float wr_s[D_EMBD * N_EXP];   // 32 KB
    for (int i = threadIdx.x; i < D_EMBD * N_EXP; i += blockDim.x) wr_s[i] = Wr[i];
    __syncthreads();

    int t = blockIdx.x * blockDim.x + threadIdx.x;
    if (t >= T_TOK) return;

    float acc[N_EXP] = {};
    const float* xr = x + (size_t)t * D_EMBD;
    for (int d = 0; d < D_EMBD; ++d) {
        float xv = xr[d];
        #pragma unroll
        for (int e = 0; e < N_EXP; ++e) acc[e] = fmaf(xv, wr_s[d * N_EXP + e], acc[e]);
    }
    float mx = -1e30f;
    #pragma unroll
    for (int e = 0; e < N_EXP; ++e) { acc[e] += br[e]; mx = fmaxf(mx, acc[e]); }
    float p[N_EXP];
    #pragma unroll
    for (int e = 0; e < N_EXP; ++e) p[e] = __expf(acc[e] - mx);

    int e0 = 0;
    #pragma unroll
    for (int e = 1; e < N_EXP; ++e) if (p[e] > p[e0]) e0 = e;
    int e1 = (e0 == 0) ? 1 : 0;
    #pragma unroll
    for (int e = 0; e < N_EXP; ++e) if (e != e0 && p[e] > p[e1]) e1 = e;

    float inv = 1.0f / (p[e0] + p[e1]);
    tok_e[t * 2 + 0] = e0;  tok_w[t * 2 + 0] = p[e0] * inv;
    tok_e[t * 2 + 1] = e1;  tok_w[t * 2 + 1] = p[e1] * inv;
    atomicAdd(&count[e0], 1);
    atomicAdd(&count[e1], 1);
}

// ---------------------------------------------------------------------------
// Exclusive prefix sum over 8 expert counts
// ---------------------------------------------------------------------------
__global__ void moe_prefix_kernel(const int* __restrict__ count,
                                  int* __restrict__ offs, int* __restrict__ cursor) {
    if (threadIdx.x == 0 && blockIdx.x == 0) {
        int a = 0;
        for (int e = 0; e < N_EXP; ++e) { offs[e] = a; cursor[e] = a; a += count[e]; }
    }
}

// ---------------------------------------------------------------------------
// Compact (token, gate) assignments into per-expert contiguous slot lists
// ---------------------------------------------------------------------------
__global__ __launch_bounds__(256)
void moe_place_kernel(const int* __restrict__ tok_e, const float* __restrict__ tok_w,
                      int* __restrict__ cursor,
                      int* __restrict__ slot_tok, float* __restrict__ slot_gate) {
    int t = blockIdx.x * blockDim.x + threadIdx.x;
    if (t >= T_TOK) return;
    #pragma unroll
    for (int k = 0; k < 2; ++k) {
        int e = tok_e[t * 2 + k];
        int s = atomicAdd(&cursor[e], 1);
        slot_tok[s]  = t;
        slot_gate[s] = tok_w[t * 2 + k];
    }
}

// ---------------------------------------------------------------------------
// Stage 1: h = gelu(X_e @ W1[e] + b1[e])  (bf16 WMMA, fp32 accum)
// grid: (FF/BN=32, 16 m-tiles, 8 experts), block 256 (8 waves)
// Software-pipelined: global loads for tile k+1 overlap WMMAs for tile k.
// ---------------------------------------------------------------------------
__global__ __launch_bounds__(256)
void moe_ffn1_kernel(const float* __restrict__ x, const float* __restrict__ W1,
                     const float* __restrict__ b1,
                     const int* __restrict__ count, const int* __restrict__ offs,
                     const int* __restrict__ slot_tok,
                     __bf16* __restrict__ h_ws) {
    const int e   = blockIdx.z;
    const int cnt = count[e];
    const int m0  = blockIdx.y * BM;
    if (m0 >= cnt) return;
    const int n0   = blockIdx.x * BN;
    const int base = offs[e];

    __shared__ __bf16 a_lds[BM * BK];     // 8 KB
    __shared__ __bf16 b_lds[BN * BKP];    // 10 KB (transposed [n][k], padded)
    __shared__ int    rows_tok[BM];

    const int tid = threadIdx.x;
    if (tid < BM) {
        int r = m0 + tid;
        rows_tok[tid] = slot_tok[base + ((r < cnt) ? r : (cnt - 1))];
    }
    __syncthreads();

    const float* W1e = W1 + (size_t)e * D_EMBD * FF_DIM;

    const int wave = tid >> 5;
    const int lane = tid & 31;
    const int lg   = lane >> 4;    // half-wave group
    const int ln   = lane & 15;

    v8f acc[BN / 16] = {};

    float4 areg[4];   // staged A tile (fp32)
    float4 breg[4];   // staged B tile (fp32)

    auto load_regs = [&](int k0) {
        #pragma unroll
        for (int i = 0; i < 4; ++i) {
            int idx = tid + i * 256;           // 0..1023
            int row = idx >> 3;
            int kk  = (idx & 7) * 4;
            areg[i] = *(const float4*)(x + (size_t)rows_tok[row] * D_EMBD + k0 + kk);
        }
        #pragma unroll
        for (int i = 0; i < 4; ++i) {
            int idx = tid + i * 256;
            int k   = idx >> 5;
            int nn  = (idx & 31) * 4;
            breg[i] = *(const float4*)(W1e + (size_t)(k0 + k) * FF_DIM + n0 + nn);
        }
    };
    auto store_lds = [&]() {
        #pragma unroll
        for (int i = 0; i < 4; ++i) {
            int idx = tid + i * 256;
            int row = idx >> 3;
            int kk  = (idx & 7) * 4;
            __bf16* dst = &a_lds[row * BK + kk];
            dst[0] = (__bf16)areg[i].x; dst[1] = (__bf16)areg[i].y;
            dst[2] = (__bf16)areg[i].z; dst[3] = (__bf16)areg[i].w;
        }
        #pragma unroll
        for (int i = 0; i < 4; ++i) {
            int idx = tid + i * 256;
            int k   = idx >> 5;
            int nn  = (idx & 31) * 4;
            b_lds[(nn + 0) * BKP + k] = (__bf16)breg[i].x;
            b_lds[(nn + 1) * BKP + k] = (__bf16)breg[i].y;
            b_lds[(nn + 2) * BKP + k] = (__bf16)breg[i].z;
            b_lds[(nn + 3) * BKP + k] = (__bf16)breg[i].w;
        }
    };

    load_regs(0);
    store_lds();
    __syncthreads();

    for (int k0 = 0; k0 < D_EMBD; k0 += BK) {
        const bool has_next = (k0 + BK) < D_EMBD;
        if (has_next) {
            __builtin_prefetch(W1e + (size_t)(k0 + 2 * BK) * FF_DIM + n0, 0, 1);
            load_regs(k0 + BK);                 // HBM latency hidden behind WMMAs
        }

        // A fragment (16x32): lanes 0-15 K 0..7 & 16..23, lanes 16-31 K 8..15 & 24..31
        AB16 af;
        {
            const __bf16* ap = &a_lds[(wave * 16 + ln) * BK + 8 * lg];
            af.h[0] = *(const v8bf*)ap;
            af.h[1] = *(const v8bf*)(ap + 16);
        }
        // Hoist ALL B fragments, then issue the 8 WMMAs back-to-back.
        AB16 bf[8];
        #pragma unroll
        for (int j = 0; j < 8; ++j) {
            const __bf16* bp = &b_lds[(j * 16 + ln) * BKP + 8 * lg];
            bf[j].h[0] = *(const v8bf*)bp;
            bf[j].h[1] = *(const v8bf*)(bp + 16);
        }
        #pragma unroll
        for (int j = 0; j < 8; ++j) {
            acc[j] = __builtin_amdgcn_wmma_f32_16x16x32_bf16(
                false, af.v, false, bf[j].v, (short)0, acc[j], false, false);
        }
        __syncthreads();                        // everyone done reading LDS
        if (has_next) {
            store_lds();                        // commit staged tile k+1
            __syncthreads();
        }
    }

    // Epilogue: +b1, exact GELU, -> bf16 h
    const int rbase = wave * 16 + 8 * lg;
    #pragma unroll
    for (int j = 0; j < 8; ++j) {
        int col = n0 + j * 16 + ln;
        float bias = b1[e * FF_DIM + col];
        #pragma unroll
        for (int r = 0; r < 8; ++r) {
            int row = rbase + r;
            if (m0 + row < cnt) {
                float v = acc[j][r] + bias;
                float g = 0.5f * v * (1.0f + erff(v * 0.70710678118654752f));
                h_ws[(size_t)(base + m0 + row) * FF_DIM + col] = (__bf16)g;
            }
        }
    }
}

// ---------------------------------------------------------------------------
// Stage 2: out[t] += gate * (h @ W2[e] + b2[e])
// grid: (D/BN=8, 16 m-tiles, 8 experts), block 256 (8 waves)
// A tile (already bf16) streamed with async global->LDS when available.
// ---------------------------------------------------------------------------
__global__ __launch_bounds__(256)
void moe_ffn2_kernel(const __bf16* __restrict__ h_ws, const float* __restrict__ W2,
                     const float* __restrict__ b2,
                     const int* __restrict__ count, const int* __restrict__ offs,
                     const int* __restrict__ slot_tok, const float* __restrict__ slot_gate,
                     float* __restrict__ out) {
    const int e   = blockIdx.z;
    const int cnt = count[e];
    const int m0  = blockIdx.y * BM;
    if (m0 >= cnt) return;
    const int n0   = blockIdx.x * BN;
    const int base = offs[e];

    __shared__ __bf16 a_lds[BM * BK];
    __shared__ __bf16 b_lds[BN * BKP];
    __shared__ int    rows_tok[BM];
    __shared__ float  rows_gate[BM];
    __shared__ int    rows_slot[BM];

    const int tid = threadIdx.x;
    if (tid < BM) {
        int r  = m0 + tid;
        int rc = (r < cnt) ? r : (cnt - 1);
        rows_slot[tid] = base + rc;
        rows_tok[tid]  = slot_tok[base + rc];
        rows_gate[tid] = slot_gate[base + rc];
    }
    __syncthreads();

    const float* W2e = W2 + (size_t)e * FF_DIM * D_EMBD;

    const int wave = tid >> 5;
    const int lane = tid & 31;
    const int lg   = lane >> 4;
    const int ln   = lane & 15;

    v8f acc[BN / 16] = {};

    float4 breg[4];   // staged B tile (fp32)

    auto load_b_regs = [&](int k0) {
        #pragma unroll
        for (int i = 0; i < 4; ++i) {
            int idx = tid + i * 256;
            int k   = idx >> 5;
            int nn  = (idx & 31) * 4;
            breg[i] = *(const float4*)(W2e + (size_t)(k0 + k) * D_EMBD + n0 + nn);
        }
    };
    auto store_b_lds = [&]() {
        #pragma unroll
        for (int i = 0; i < 4; ++i) {
            int idx = tid + i * 256;
            int k   = idx >> 5;
            int nn  = (idx & 31) * 4;
            b_lds[(nn + 0) * BKP + k] = (__bf16)breg[i].x;
            b_lds[(nn + 1) * BKP + k] = (__bf16)breg[i].y;
            b_lds[(nn + 2) * BKP + k] = (__bf16)breg[i].z;
            b_lds[(nn + 3) * BKP + k] = (__bf16)breg[i].w;
        }
    };
    // A tile: h is already bf16 -> direct global->LDS copy (async if supported)
    auto a_copy = [&](int k0) {
        #pragma unroll
        for (int i = 0; i < 2; ++i) {
            int idx = tid + i * 256;           // 0..511 chunks of 16B
            int row = idx >> 2;
            int c   = (idx & 3) * 8;           // bf16 elems
            __bf16* src = const_cast<__bf16*>(h_ws) +
                          (size_t)rows_slot[row] * FF_DIM + k0 + c;
            __bf16* dst = &a_lds[row * BK + c];
#if MOE_ASYNC_LDS
            __builtin_amdgcn_global_load_async_to_lds_b128(
                (as1_v4i*)src, (as3_v4i*)dst, 0, 0);
#else
            *(uint4*)dst = *(const uint4*)src;
#endif
        }
    };

    load_b_regs(0);
    a_copy(0);
    store_b_lds();
    moe_wait_async();
    __syncthreads();

    for (int k0 = 0; k0 < FF_DIM; k0 += BK) {
        const bool has_next = (k0 + BK) < FF_DIM;
        if (has_next) {
            __builtin_prefetch(W2e + (size_t)(k0 + 2 * BK) * D_EMBD + n0, 0, 1);
            load_b_regs(k0 + BK);
        }

        AB16 af;
        {
            const __bf16* ap = &a_lds[(wave * 16 + ln) * BK + 8 * lg];
            af.h[0] = *(const v8bf*)ap;
            af.h[1] = *(const v8bf*)(ap + 16);
        }
        AB16 bf[8];
        #pragma unroll
        for (int j = 0; j < 8; ++j) {
            const __bf16* bp = &b_lds[(j * 16 + ln) * BKP + 8 * lg];
            bf[j].h[0] = *(const v8bf*)bp;
            bf[j].h[1] = *(const v8bf*)(bp + 16);
        }
        #pragma unroll
        for (int j = 0; j < 8; ++j) {
            acc[j] = __builtin_amdgcn_wmma_f32_16x16x32_bf16(
                false, af.v, false, bf[j].v, (short)0, acc[j], false, false);
        }
        __syncthreads();
        if (has_next) {
            a_copy(k0 + BK);
            store_b_lds();
            moe_wait_async();
            __syncthreads();
        }
    }

    // Epilogue: (acc + b2) * gate, atomic combine into out (2 adds/elem total)
    const int rbase = wave * 16 + 8 * lg;
    #pragma unroll
    for (int j = 0; j < 8; ++j) {
        int col = n0 + j * 16 + ln;
        float bias = b2[e * D_EMBD + col];
        #pragma unroll
        for (int r = 0; r < 8; ++r) {
            int row = rbase + r;
            if (m0 + row < cnt) {
                float val = (acc[j][r] + bias) * rows_gate[row];
                atomicAdd(&out[(size_t)rows_tok[row] * D_EMBD + col], val);
            }
        }
    }
}

// ---------------------------------------------------------------------------
// Launch
// ---------------------------------------------------------------------------
extern "C" void kernel_launch(void* const* d_in, const int* in_sizes, int n_in,
                              void* d_out, int out_size, void* d_ws, size_t ws_size,
                              hipStream_t stream) {
    const float* x  = (const float*)d_in[0];
    const float* Wr = (const float*)d_in[1];
    const float* br = (const float*)d_in[2];
    const float* W1 = (const float*)d_in[3];
    const float* b1 = (const float*)d_in[4];
    const float* W2 = (const float*)d_in[5];
    const float* b2 = (const float*)d_in[6];
    float* out = (float*)d_out;

    // Workspace layout
    char* ws = (char*)d_ws;
    __bf16* h_ws = (__bf16*)ws;                       // [2T, FF] bf16 = 32 MB
    size_t p = (size_t)2 * T_TOK * FF_DIM * sizeof(__bf16);
    int*   tok_e     = (int*)(ws + p);   p += (size_t)T_TOK * 2 * sizeof(int);
    float* tok_w     = (float*)(ws + p); p += (size_t)T_TOK * 2 * sizeof(float);
    int*   slot_tok  = (int*)(ws + p);   p += (size_t)T_TOK * 2 * sizeof(int);
    float* slot_gate = (float*)(ws + p); p += (size_t)T_TOK * 2 * sizeof(float);
    int*   count     = (int*)(ws + p);   p += N_EXP * sizeof(int);
    int*   offs      = (int*)(ws + p);   p += N_EXP * sizeof(int);
    int*   cursor    = (int*)(ws + p);   p += N_EXP * sizeof(int);

    // 1) zero out + counters
    moe_zero_kernel<<<(T_TOK * D_EMBD + 255) / 256, 256, 0, stream>>>(out, count);
    // 2) router
    moe_router_kernel<<<T_TOK / 256, 256, 0, stream>>>(x, Wr, br, tok_e, tok_w, count);
    // 3) prefix
    moe_prefix_kernel<<<1, 32, 0, stream>>>(count, offs, cursor);
    // 4) placement
    moe_place_kernel<<<T_TOK / 256, 256, 0, stream>>>(tok_e, tok_w, cursor, slot_tok, slot_gate);
    // 5) stage-1 grouped GEMM + GELU
    moe_ffn1_kernel<<<dim3(FF_DIM / BN, T_TOK / BM, N_EXP), 256, 0, stream>>>(
        x, W1, b1, count, offs, slot_tok, h_ws);
    // 6) stage-2 grouped GEMM + weighted combine
    moe_ffn2_kernel<<<dim3(D_EMBD / BN, T_TOK / BM, N_EXP), 256, 0, stream>>>(
        h_ws, W2, b2, count, offs, slot_tok, slot_gate, out);
}
